// SkipEmbedder_87935160418865
// MI455X (gfx1250) — matrix operand order repaired and checked by reading, not compiled, not run on hardware
//
#include <hip/hip_runtime.h>
#include <cstdint>
#include <cstddef>

#define NN 128   // nodes
#define HH 128   // hidden
#define TT 32    // steps
#define G4 512   // 4*H gate width

typedef __attribute__((ext_vector_type(16))) __bf16       v16bf;
typedef __attribute__((ext_vector_type(8)))  float        v8f;
typedef __attribute__((ext_vector_type(8)))  unsigned int v8u;
typedef __attribute__((ext_vector_type(4)))  unsigned int v4u;

__device__ __forceinline__ float rcpf(float x)  { return __builtin_amdgcn_rcpf(x); }
__device__ __forceinline__ float rsqf(float x)  { return __builtin_amdgcn_rsqf(x); }
__device__ __forceinline__ float sigf(float x)  { return rcpf(1.0f + __expf(-x)); }
__device__ __forceinline__ float tanhfast(float x) {
  x = fminf(fmaxf(x, -15.0f), 15.0f);
  float e = __expf(2.0f * x);
  return fmaf(-2.0f, rcpf(e + 1.0f), 1.0f);
}
__device__ __forceinline__ unsigned pack_bf16(float a, float b) {
  return (__float_as_uint(a) >> 16) | (__float_as_uint(b) & 0xffff0000u);
}
__device__ __forceinline__ float bf_lo(unsigned u) { return __uint_as_float(u << 16); }
__device__ __forceinline__ float bf_hi(unsigned u) { return __uint_as_float(u & 0xffff0000u); }
__device__ __forceinline__ v8u ldsv8(const unsigned* p) {
  return *(const v8u*)__builtin_assume_aligned((const void*)p, 16);
}

// ---------------------------------------------------------------------------
// Prep 1: X = E @ Wi + b in WMMA C-fragment order.
// ---------------------------------------------------------------------------
extern "C" __global__ void se_prep_x(const float* __restrict__ E,
                                     const float* __restrict__ Wi,
                                     const float* __restrict__ b,
                                     float* __restrict__ Xf) {
  int n = blockIdx.x;
  int j = threadIdx.x;     // 512 threads
  float s = b[j];
  #pragma unroll 8
  for (int k = 0; k < HH; ++k) s = fmaf(E[n * HH + k], Wi[k * G4 + j], s);
  int mt = n >> 4, rr = n & 15, r = rr & 7, hi = rr >> 3;
  int nt = j >> 4, l = hi * 16 + (j & 15);
  Xf[(mt * 32 + nt) * 256 + l * 8 + r] = s;
}

// ---------------------------------------------------------------------------
// Prep 2: CSR of incoming edges (static across steps/sources)
// ---------------------------------------------------------------------------
extern "C" __global__ void se_prep_csr(const int* __restrict__ ti,
                                       const int* __restrict__ fi,
                                       int* __restrict__ roff,
                                       int* __restrict__ esrc,
                                       int* __restrict__ ebr) {
  int cnt[NN];
  for (int i = 0; i < NN; ++i) cnt[i] = 0;
  for (int n = 0; n < NN; ++n) { cnt[ti[n]]++; cnt[fi[n]]++; }
  int run = 0;
  for (int i = 0; i < NN; ++i) { roff[i] = run; run += cnt[i]; cnt[i] = 0; }
  roff[NN] = run;
  for (int n = 0; n < NN; ++n) {
    int m = ti[n]; int p = roff[m] + cnt[m]++; esrc[p] = n; ebr[p] = 0;
    m = fi[n];     p = roff[m] + cnt[m]++;     esrc[p] = n; ebr[p] = 1;
  }
}

// ---------------------------------------------------------------------------
// Main: one workgroup per source, 8 waves; wave w owns rows [16w,16w+16).
// gates = h@Wh (128 WMMA/wave/step); c'=S@c, h'=S@h (64 WMMA/wave/step).
// c-state: bf16 row-pair LDS buffer; acc: registers; h-state: bf16 A-frags.
// ---------------------------------------------------------------------------
extern "C" __global__ __launch_bounds__(256, 1)
void se_main(const float* __restrict__ E,  const float* __restrict__ WhG,
             const float* __restrict__ Wd, const float* __restrict__ bd,
             const float* __restrict__ gma, const float* __restrict__ bta,
             const float* __restrict__ Xf, const int* __restrict__ roff,
             const int* __restrict__ esrc, const int* __restrict__ ebr,
             float* __restrict__ out) {
  extern __shared__ char smem[];
  unsigned* lds_whB = (unsigned*)smem;        // Wh bf16 B-frags    32768 dw 128KB
  unsigned* lds_cB  = lds_whB + 32768;        // c_new bf16 B-frags  8192 dw 32KB
  unsigned* lds_hB  = lds_cB + 8192;          // h_new bf16 B-frags  8192 dw 32KB
  unsigned* lds_S   = lds_hB + 8192;          // S bf16 pairs row-major      32KB
  unsigned* lds_hA  = lds_S + 8192;           // h_agg bf16 A-frags          32KB
  unsigned* lds_cS  = lds_hA + 8192;          // c-state bf16 row pairs      32KB
  float* fsm       = (float*)(lds_cS + 8192);
  float* lds_ip    = fsm;                     // 128
  float* lds_wt    = fsm + 128;               // 128
  float* lds_wf    = fsm + 256;               // 128
  float* lds_logit = fsm + 384;               // 256
  float* lds_wd    = fsm + 640;               // 512
  float* lds_bd    = fsm + 1152;              // 2 (+pad)
  int*   lds_roff  = (int*)(fsm + 1156);      // 129 (+pad)
  int*   lds_esrc  = (int*)(fsm + 1288);      // 256
  int*   lds_ebr   = (int*)(fsm + 1544);      // 256

  const int tid  = threadIdx.x;
  const int lane = tid & 31;
  const int wv   = tid >> 5;
  const int hi   = lane >> 4;
  const int src  = blockIdx.x;
  const int rowbase = (wv << 4) + (hi << 3);  // first of this lane's 8 rows
  const int prbase  = (rowbase >> 1);         // row-pair index base

  // ---- Wh -> LDS bf16 B-fragments -----------------------------------------
  for (int f = wv; f < 128; f += 8) {
    int kt = f >> 5, nt = f & 31;
    int n  = (nt << 4) + (lane & 15);
    int kb = (kt << 5) + (hi << 3);
    unsigned* dst = lds_whB + (((size_t)f * 32 + lane) << 3);
    #pragma unroll
    for (int r = 0; r < 8; ++r) {
      int k0 = kb + 2 * r + ((r >> 2) << 3);
      dst[r] = pack_bf16(WhG[k0 * G4 + n], WhG[(k0 + 1) * G4 + n]);
    }
  }
  // ---- init (h0 = 0, c0 = 0) ----------------------------------------------
  for (int i = tid; i < 2048; i += 256) {
    ((v4u*)lds_hA)[i] = (v4u){0u, 0u, 0u, 0u};
    ((v4u*)lds_cS)[i] = (v4u){0u, 0u, 0u, 0u};
  }
  if (tid < NN) lds_ip[tid] = (tid == src) ? 1.0f : 0.0f;
  for (int i = tid; i < 512; i += 256) lds_wd[i] = Wd[i];
  if (tid < 2) lds_bd[tid] = bd[tid];
  for (int i = tid; i < NN + 1; i += 256) lds_roff[i] = roff[i];
  lds_esrc[tid] = esrc[tid];
  lds_ebr[tid]  = ebr[tid];

  v8f accT[8];     // running sum of h_agg*ip (C-tile layout)
  float hlast[8];  // previous h_agg (row-127 lanes only)
  #pragma unroll
  for (int dt = 0; dt < 8; ++dt) { accT[dt] = {}; hlast[dt] = 0.0f; }
  __syncthreads();

  for (int step = 0; step < TT; ++step) {
    // ---- phase 1: A-fragments of h (bf16, direct vector loads) ------------
    v8u aU[4];
    #pragma unroll
    for (int kt = 0; kt < 4; ++kt)
      aU[kt] = ldsv8(lds_hA + ((((size_t)wv * 4 + kt) * 32 + lane) << 3));

    // ---- phase 2: gates WMMA + LSTM elementwise + operand emission --------
    float lg0[8], lg1[8];
    #pragma unroll
    for (int e = 0; e < 8; ++e) { lg0[e] = 0.0f; lg1[e] = 0.0f; }
    const int laneP = (lane & 15) + ((wv & 1) << 4);
    const int kt2   = wv >> 1;
    #pragma unroll
    for (int dt = 0; dt < 8; ++dt) {
      v8f g4[4];
      #pragma unroll
      for (int g = 0; g < 4; ++g) {
        const float* xp = Xf + (((size_t)wv * 32 + (g * 8 + dt)) * 256 + lane * 8);
        #pragma unroll
        for (int r = 0; r < 8; ++r) g4[g][r] = xp[r];
      }
      #pragma unroll
      for (int kt = 0; kt < 4; ++kt) {
        v16bf aF = __builtin_bit_cast(v16bf, aU[kt]);
        v8u b0 = ldsv8(lds_whB + ((((size_t)kt * 32 + (0 * 8 + dt)) * 32 + lane) << 3));
        v8u b1 = ldsv8(lds_whB + ((((size_t)kt * 32 + (1 * 8 + dt)) * 32 + lane) << 3));
        v8u b2 = ldsv8(lds_whB + ((((size_t)kt * 32 + (2 * 8 + dt)) * 32 + lane) << 3));
        v8u b3 = ldsv8(lds_whB + ((((size_t)kt * 32 + (3 * 8 + dt)) * 32 + lane) << 3));
        g4[0] = __builtin_amdgcn_wmma_f32_16x16x32_bf16(
            false, aF, false, __builtin_bit_cast(v16bf, b0), (short)0, g4[0], false, false);
        g4[1] = __builtin_amdgcn_wmma_f32_16x16x32_bf16(
            false, aF, false, __builtin_bit_cast(v16bf, b1), (short)0, g4[1], false, false);
        g4[2] = __builtin_amdgcn_wmma_f32_16x16x32_bf16(
            false, aF, false, __builtin_bit_cast(v16bf, b2), (short)0, g4[2], false, false);
        g4[3] = __builtin_amdgcn_wmma_f32_16x16x32_bf16(
            false, aF, false, __builtin_bit_cast(v16bf, b3), (short)0, g4[3], false, false);
      }
      int d = (dt << 4) + (lane & 15);
      float wd0 = lds_wd[d * 2],        wd1 = lds_wd[d * 2 + 1];
      float wd2 = lds_wd[(HH + d) * 2], wd3 = lds_wd[(HH + d) * 2 + 1];
      // c-state: 4 bf16 row-pair dwords for this lane's 8 rows at column d
      unsigned cpair[4];
      #pragma unroll
      for (int p = 0; p < 4; ++p) cpair[p] = lds_cS[(unsigned)(prbase + p) * HH + d];
      float cne[8], hne[8];
      #pragma unroll
      for (int e = 0; e < 8; ++e) {
        float co = (e & 1) ? bf_hi(cpair[e >> 1]) : bf_lo(cpair[e >> 1]);
        float ig = sigf(g4[0][e]);
        float fg = sigf(g4[1][e]);
        float gg = tanhfast(g4[2][e]);
        float og = sigf(g4[3][e]);
        float cn = fmaf(fg, co, ig * gg);
        float hn = og * tanhfast(cn);
        int row = rowbase + e;
        if (row == NN - 1) { cn = co; hn = hlast[dt]; }   // exit node holds
        cne[e] = cn; hne[e] = hn;
        lg0[e] += cn * wd0 + hn * wd2;
        lg1[e] += cn * wd1 + hn * wd3;
      }
      // bf16 B-fragment copies for aggregation (lane-local row pairs)
      unsigned* basec = lds_cB + ((((size_t)kt2 * 8 + dt) * 32 + laneP) << 3);
      unsigned* baseh = lds_hB + ((((size_t)kt2 * 8 + dt) * 32 + laneP) << 3);
      #pragma unroll
      for (int ee = 0; ee < 8; ee += 2) {
        int rp = (hi << 2) + (ee >> 1);
        basec[rp] = pack_bf16(cne[ee], cne[ee + 1]);
        baseh[rp] = pack_bf16(hne[ee], hne[ee + 1]);
      }
    }
    // ---- phase 3: logits reduce over 16-lane half -------------------------
    #pragma unroll
    for (int e = 0; e < 8; ++e) {
      #pragma unroll
      for (int off = 1; off < 16; off <<= 1) {
        lg0[e] += __shfl_xor(lg0[e], off, 32);
        lg1[e] += __shfl_xor(lg1[e], off, 32);
      }
    }
    if ((lane & 15) < 2) {
      int lgi = lane & 15;
      #pragma unroll
      for (int e = 0; e < 8; ++e)
        lds_logit[(rowbase + e) * 2 + lgi] = lgi ? lg1[e] : lg0[e];
    }
    __syncthreads();

    // ---- phase 4: softmax -> branch weights; zero S (b128 stores) ---------
    if (tid < NN) {
      float l0 = lds_logit[2 * tid] + lds_bd[0];
      float l1 = lds_logit[2 * tid + 1] + lds_bd[1];
      float mx = fmaxf(l0, l1);
      float e0 = __expf(l0 - mx), e1 = __expf(l1 - mx);
      float inv = rcpf(e0 + e1);
      float ipv = lds_ip[tid];
      lds_wt[tid] = e0 * inv * ipv;
      lds_wf[tid] = e1 * inv * ipv;
    }
    for (int i = tid; i < 2048; i += 256) ((v4u*)lds_S)[i] = (v4u){0u, 0u, 0u, 0u};
    __syncthreads();

    // ---- phase 5: ip_new + build S (bf16 pairs, row-major) ----------------
    if (tid < NN) {
      int m = tid; float s = 0.0f;
      for (int e = lds_roff[m]; e < lds_roff[m + 1]; ++e) {
        int n2 = lds_esrc[e];
        float w = lds_ebr[e] ? lds_wf[n2] : lds_wt[n2];
        s += w;
        unsigned idx = (unsigned)m * 64u + (unsigned)(n2 >> 1);
        unsigned u = lds_S[idx];
        if (n2 & 1) {
          float prev = __uint_as_float(u & 0xffff0000u);
          lds_S[idx] = (u & 0x0000ffffu) | (__float_as_uint(prev + w) & 0xffff0000u);
        } else {
          float prev = __uint_as_float(u << 16);
          lds_S[idx] = (u & 0xffff0000u) | (__float_as_uint(prev + w) >> 16);
        }
      }
      lds_ip[tid] = s;   // ip_new
    }
    __syncthreads();

    // ---- phase 6: aggregation via WMMA (4 independent chains) -------------
    v8u aS[4];
    {
      int m = (wv << 4) + (lane & 15);
      #pragma unroll
      for (int kt = 0; kt < 4; ++kt) {
        const v4u* p = (const v4u*)__builtin_assume_aligned(
            (const void*)(lds_S + (unsigned)m * 64u + (kt << 4) + (hi << 3)), 16);
        v4u lo = p[0], hp = p[1];
        aS[kt] = (v8u){lo.x, lo.y, lo.z, lo.w, hp.x, hp.y, hp.z, hp.w};
      }
    }
    float rden[8], wacc[8];
    #pragma unroll
    for (int r = 0; r < 8; ++r) {
      float ipn = lds_ip[rowbase + r];
      float rd = rcpf(ipn + 1e-7f);
      rden[r] = rd; wacc[r] = ipn * rd;
    }
    unsigned short* hA16 = (unsigned short*)lds_hA;
    #pragma unroll
    for (int dt = 0; dt < 8; ++dt) {
      v8f cc0 = {}, cc1 = {}, hh0 = {}, hh1 = {};
      #pragma unroll
      for (int kk = 0; kk < 2; ++kk) {          // kt pairs (2k, 2k+1)
        int ktA = 2 * kk, ktB = 2 * kk + 1;
        v8u bc0 = ldsv8(lds_cB + ((((size_t)ktA * 8 + dt) * 32 + lane) << 3));
        v8u bh0 = ldsv8(lds_hB + ((((size_t)ktA * 8 + dt) * 32 + lane) << 3));
        v8u bc1 = ldsv8(lds_cB + ((((size_t)ktB * 8 + dt) * 32 + lane) << 3));
        v8u bh1 = ldsv8(lds_hB + ((((size_t)ktB * 8 + dt) * 32 + lane) << 3));
        v16bf a0 = __builtin_bit_cast(v16bf, aS[ktA]);
        v16bf a1 = __builtin_bit_cast(v16bf, aS[ktB]);
        cc0 = __builtin_amdgcn_wmma_f32_16x16x32_bf16(
            false, a0, false, __builtin_bit_cast(v16bf, bc0), (short)0, cc0, false, false);
        hh0 = __builtin_amdgcn_wmma_f32_16x16x32_bf16(
            false, a0, false, __builtin_bit_cast(v16bf, bh0), (short)0, hh0, false, false);
        cc1 = __builtin_amdgcn_wmma_f32_16x16x32_bf16(
            false, a1, false, __builtin_bit_cast(v16bf, bc1), (short)0, cc1, false, false);
        hh1 = __builtin_amdgcn_wmma_f32_16x16x32_bf16(
            false, a1, false, __builtin_bit_cast(v16bf, bh1), (short)0, hh1, false, false);
      }
      // scale, commit c-state (bf16 row pairs), acc, emit next A operand
      int d   = (dt << 4) + (lane & 15);
      int kp  = d & 31;
      int kt3 = dt >> 1;
      int laneA_base = 16 * ((kp >> 3) & 1);
      int rA = ((dt & 1) << 2) + ((lane & 7) >> 1);
      float ca[8], ha[8];
      #pragma unroll
      for (int r = 0; r < 8; ++r) {
        float cs = cc0[r] + cc1[r];
        float hs = hh0[r] + hh1[r];
        ca[r] = cs * rden[r];
        ha[r] = hs * rden[r];
        accT[dt][r] += hs * wacc[r];
        if (r == 7) hlast[dt] = ha[7];
        int laneA = (r + (hi << 3)) + laneA_base;
        unsigned dwidx = (((unsigned)wv * 4 + kt3) * 32 + laneA) * 8 + rA;
        hA16[dwidx * 2 + (lane & 1)] = (unsigned short)(__float_as_uint(ha[r]) >> 16);
      }
      #pragma unroll
      for (int r = 0; r < 8; r += 2)
        lds_cS[(unsigned)(prbase + (r >> 1)) * HH + d] = pack_bf16(ca[r], ca[r + 1]);
    }
    __syncthreads();   // protect cB/hB/S/cS against next step's writers
  }

  // ---- layernorm + writeout straight from registers -------------------------
  #pragma unroll
  for (int r = 0; r < 8; ++r) {
    float s = 0.0f, s2 = 0.0f;
    #pragma unroll
    for (int dt = 0; dt < 8; ++dt) { float v = accT[dt][r]; s += v; s2 += v * v; }
    #pragma unroll
    for (int off = 1; off < 16; off <<= 1) {
      s  += __shfl_xor(s,  off, 32);
      s2 += __shfl_xor(s2, off, 32);
    }
    float mu   = s * (1.0f / HH);
    float var  = s2 * (1.0f / HH) - mu * mu;
    float rstd = rsqf(var + 1e-6f);
    int row = rowbase + r;
    #pragma unroll
    for (int dt = 0; dt < 8; ++dt) {
      int d = (dt << 4) + (lane & 15);
      float v = (row == src) ? E[row * HH + d]
                             : (accT[dt][r] - mu) * rstd * gma[d] + bta[d];
      out[((size_t)src * NN + row) * HH + d] = v;
    }
  }
}

// ---------------------------------------------------------------------------
extern "C" void kernel_launch(void* const* d_in, const int* in_sizes, int n_in,
                              void* d_out, int out_size, void* d_ws, size_t ws_size,
                              hipStream_t stream) {
  const float* E    = (const float*)d_in[0];
  const int*   ti   = (const int*)d_in[1];
  const int*   fi   = (const int*)d_in[2];
  const float* Wi   = (const float*)d_in[3];
  const float* Wh   = (const float*)d_in[4];
  const float* b    = (const float*)d_in[5];
  const float* Wd   = (const float*)d_in[6];
  const float* bd   = (const float*)d_in[7];
  const float* gma  = (const float*)d_in[8];
  const float* bta  = (const float*)d_in[9];

  char*  ws   = (char*)d_ws;
  float* Xf   = (float*)ws;                          // 65536 f32 = 256KB
  int*   roff = (int*)(ws + 65536 * sizeof(float));  // 129 (+pad)
  int*   esrc = roff + 132;                          // 256
  int*   ebr  = esrc + 256;                          // 256

  se_prep_x<<<dim3(NN), dim3(G4), 0, stream>>>(E, Wi, b, Xf);
  se_prep_csr<<<dim3(1), dim3(1), 0, stream>>>(ti, fi, roff, esrc, ebr);

  constexpr size_t SMEM = (size_t)(32768 + 5 * 8192) * 4  // whB+cB+hB+S+hA+cS
                        + 8192;                           // small tables
  hipFuncSetAttribute(reinterpret_cast<const void*>(se_main),
                      hipFuncAttributeMaxDynamicSharedMemorySize, (int)SMEM);
  se_main<<<dim3(NN), dim3(256), SMEM, stream>>>(
      E, Wh, Wd, bd, gma, bta, Xf, roff, esrc, ebr, (float*)d_out);
}
